// VariationalPrototypeLearning_52424370815316
// MI455X (gfx1250) — compile-verified, matrix operand order checked by reading, not run on hardware
//
#include <hip/hip_runtime.h>

// ---- types -----------------------------------------------------------------
typedef __bf16        v16bf  __attribute__((ext_vector_type(16)));
typedef float         v8f    __attribute__((ext_vector_type(8)));
typedef unsigned int  u32x4  __attribute__((ext_vector_type(4)));
typedef float         f32x4  __attribute__((ext_vector_type(4)));

#define BROWS   512
#define DDIM    256
#define CDIM    100000
#define NCHUNKS 50
#define CHUNK_N 2000          // 125 tiles of 16; 50*2000 == 100000 exactly
#define NTILES  (CHUNK_N/16)  // 125
#define NPART   (NCHUNKS*2)   // two waves (n-parity) per chunk write partials

#define SSCALE  64.0f         // also the fixed softmax shift: logits <= 64
#define COS_M   0.8775825618903728f   // cos(0.5)
#define SIN_M   0.4794255386042030f   // sin(0.5)

__device__ __forceinline__ unsigned short f2bf(float f) {
  unsigned int u = __float_as_uint(f);
  unsigned int r = u + 0x7FFFu + ((u >> 16) & 1u);   // round-to-nearest-even
  return (unsigned short)(r >> 16);
}

// ---- kernel 1/2: row-normalize (nrows x 256 f32) -> bf16, 1 wave per row ---
__global__ void rownorm_to_bf16(const float* __restrict__ src,
                                unsigned short* __restrict__ dst) {
  const int row  = blockIdx.x;
  const int lane = threadIdx.x;                 // 0..31
  const float* p = src + (size_t)row * DDIM + lane * 8;
  f32x4 a = *(const f32x4*)(p);
  f32x4 b = *(const f32x4*)(p + 4);
  float ss = a.x*a.x + a.y*a.y + a.z*a.z + a.w*a.w
           + b.x*b.x + b.y*b.y + b.z*b.z + b.w*b.w;
  #pragma unroll
  for (int m = 16; m >= 1; m >>= 1) ss += __shfl_xor(ss, m, 32);
  const float scale = 1.0f / fmaxf(sqrtf(ss), 1e-12f);

  unsigned int h0 = (unsigned int)f2bf(a.x*scale) | ((unsigned int)f2bf(a.y*scale) << 16);
  unsigned int h1 = (unsigned int)f2bf(a.z*scale) | ((unsigned int)f2bf(a.w*scale) << 16);
  unsigned int h2 = (unsigned int)f2bf(b.x*scale) | ((unsigned int)f2bf(b.y*scale) << 16);
  unsigned int h3 = (unsigned int)f2bf(b.z*scale) | ((unsigned int)f2bf(b.w*scale) << 16);
  u32x4 o; o.x = h0; o.y = h1; o.z = h2; o.w = h3;
  *(u32x4*)(dst + (size_t)row * DDIM + lane * 8) = o;
}

// ---- kernel 3: fused WMMA GEMM + fixed-shift softmax partials --------------
// grid = (NCHUNKS, BROWS/64); block = 256 (8 waves).
// A-block staged via LDS (compiler promotes the 8 fragments to VGPRs).
// The FULL next B tile (8 k-steps, 64 VGPRs) is prefetched right after the
// current tile's WMMAs consume it, so the 8 v_wmma run back-to-back and the
// epilogue + next WMMA burst hide the L2 latency of the 16 in-flight loads.
__global__ void __launch_bounds__(256)
arc_gemm_softmax(const unsigned short* __restrict__ Ebf,
                 const unsigned short* __restrict__ Pbf,
                 const int* __restrict__ labels,
                 float* __restrict__ spart,
                 float* __restrict__ tgt) {
  __shared__ unsigned short Alds[64 * DDIM];        // 32 KB

  const int chunk  = blockIdx.x;
  const int mblk   = blockIdx.y;
  const int w      = threadIdx.x >> 5;
  const int lane   = threadIdx.x & 31;
  const int mt     = w >> 1;
  const int parity = w & 1;
  const int lg     = lane & 15;       // N index / A-row index within tile
  const int hi     = lane >> 4;       // K-group select per ISA 16-bit layout
  const int mbase  = mblk * 64 + mt * 16;

  // Cooperative A-block copy: 256 threads x 128B = 32KB.
  {
    const unsigned short* src = Ebf + (size_t)mblk * 64 * DDIM + threadIdx.x * 64;
    unsigned short* dstl = Alds + threadIdx.x * 64;
    #pragma unroll
    for (int i = 0; i < 8; ++i)
      ((u32x4*)dstl)[i] = ((const u32x4*)src)[i];
  }
  __syncthreads();

  const unsigned short* Arow = Alds + (mt * 16 + lg) * DDIM;  // this lane's A row

  // Rows owned by this lane-half in the C/D layout: M = j + 8*hi.
  int   labs[8];
  float rs[8];                     // per-lane partial sum of exp(l - 64)
  #pragma unroll
  for (int j = 0; j < 8; ++j) {
    labs[j] = labels[mbase + hi * 8 + j];
    rs[j]   = 0.0f;
  }

  const int nbase0 = chunk * CHUNK_N;
  typedef union { u32x4 q[2]; v16bf v; } Frag;

  // Prologue: load the ENTIRE first B tile (8 k-steps, 64 VGPRs).
  int ncol = nbase0 + parity * 16 + lg;
  const unsigned short* brow = Pbf + (size_t)ncol * DDIM;
  Frag bt[8];
  #pragma unroll
  for (int ks = 0; ks < 8; ++ks) {
    const int k0 = ks * 32 + hi * 8;
    bt[ks].q[0] = *(const u32x4*)(brow + k0);
    bt[ks].q[1] = *(const u32x4*)(brow + k0 + 16);
  }

  for (int nt = parity; nt < NTILES; nt += 2) {
    // 8 back-to-back WMMAs: all operands register-resident.
    v8f acc = {0.f, 0.f, 0.f, 0.f, 0.f, 0.f, 0.f, 0.f};
    #pragma unroll
    for (int ks = 0; ks < 8; ++ks) {
      Frag a;
      const int ka = ks * 32 + hi * 8;
      a.q[0] = *(const u32x4*)(Arow + ka);
      a.q[1] = *(const u32x4*)(Arow + ka + 16);
      acc = __builtin_amdgcn_wmma_f32_16x16x32_bf16(
              false, a.v, false, bt[ks].v, (short)0, acc, false, false);
    }

    const int curcol = ncol;
    // bt[] fully consumed: immediately issue the whole next tile's loads so
    // they fly during the epilogue and the next tile's WMMA burst.
    if (nt + 2 < NTILES) {
      ncol = nbase0 + (nt + 2) * 16 + lg;
      brow = Pbf + (size_t)ncol * DDIM;
      #pragma unroll
      for (int ks = 0; ks < 8; ++ks) {
        const int k0 = ks * 32 + hi * 8;
        bt[ks].q[0] = *(const u32x4*)(brow + k0);
        bt[ks].q[1] = *(const u32x4*)(brow + k0 + 16);
      }
    }

    // Epilogue: fixed-shift softmax accumulation, per-lane (no shuffles).
    // Margin substitution only on the (rare) target column.
    #pragma unroll
    for (int j = 0; j < 8; ++j) {
      const float c = acc[j];
      float lsh = fmaf(SSCALE, c, -SSCALE);          // logit - 64
      if (curcol == labs[j]) {
        float sine = sqrtf(fmaxf(1.0f - c * c, 1e-16f));
        sine = fminf(fmaxf(sine, 1e-16f), 1.0f - 1e-16f);
        const float lt = SSCALE * (c * COS_M - sine * SIN_M);
        tgt[mbase + hi * 8 + j] = lt;                // unique writer per row
        lsh = lt - SSCALE;
      }
      rs[j] += __expf(lsh);
    }
  }

  // One-time cross-lane merge within each 16-lane half, then write partials.
  #pragma unroll
  for (int j = 0; j < 8; ++j) {
    float s = rs[j];
    s += __shfl_xor(s, 1, 32);
    s += __shfl_xor(s, 2, 32);
    s += __shfl_xor(s, 4, 32);
    s += __shfl_xor(s, 8, 32);
    rs[j] = s;
  }
  if (lg == 0) {                                     // lanes 0 and 16 write
    const int p = chunk * 2 + parity;
    #pragma unroll
    for (int j = 0; j < 8; ++j)
      spart[p * BROWS + (mbase + hi * 8 + j)] = rs[j];
  }
}

// ---- kernel 4: merge partials + mean loss ----------------------------------
__global__ void arc_finalize(const float* __restrict__ spart,
                             const float* __restrict__ tgt,
                             float* __restrict__ out) {
  __shared__ float red[256];
  const int tid = threadIdx.x;
  float acc = 0.0f;
  for (int r = tid; r < BROWS; r += 256) {
    float S = 0.0f;
    for (int p = 0; p < NPART; ++p) S += spart[p * BROWS + r];
    acc += (SSCALE + __logf(S)) - tgt[r];            // logsumexp - target logit
  }
  red[tid] = acc;
  __syncthreads();
  for (int s = 128; s >= 1; s >>= 1) {
    if (tid < s) red[tid] += red[tid + s];
    __syncthreads();
  }
  if (tid == 0) {
    out[0] = red[0] / (float)BROWS;
    out[1] = -1.0f;
    out[2] = -1.0f;
  }
}

// ---- launcher --------------------------------------------------------------
extern "C" void kernel_launch(void* const* d_in, const int* in_sizes, int n_in,
                              void* d_out, int out_size, void* d_ws, size_t ws_size,
                              hipStream_t stream) {
  const float* emb    = (const float*)d_in[0];   // (512, 256) f32
  const int*   labels = (const int*)  d_in[1];   // (512,)
  const float* proto  = (const float*)d_in[2];   // (100000, 256) f32

  char* ws = (char*)d_ws;
  unsigned short* Ebf = (unsigned short*)ws;                       // 256 KB
  unsigned short* Pbf = (unsigned short*)(ws + (size_t)BROWS * DDIM * 2);
  char* ws2   = ws + (size_t)BROWS * DDIM * 2 + (size_t)CDIM * DDIM * 2;
  float* tgt  = (float*)ws2;                                       // 512 f32
  float* sprt = (float*)(ws2 + BROWS * 4);                         // 100x512 f32

  rownorm_to_bf16<<<BROWS, 32, 0, stream>>>(emb, Ebf);
  rownorm_to_bf16<<<CDIM, 32, 0, stream>>>(proto, Pbf);

  dim3 grid(NCHUNKS, BROWS / 64);
  arc_gemm_softmax<<<grid, 256, 0, stream>>>(Ebf, Pbf, labels, sprt, tgt);

  arc_finalize<<<1, 256, 0, stream>>>(sprt, tgt, (float*)d_out);
}
